// Attention_6313601925665
// MI455X (gfx1250) — matrix operand order repaired and checked by reading, not compiled, not run on hardware
//
#include <hip/hip_runtime.h>
#include <hip/hip_bf16.h>
#include <math.h>

// ---------------------------------------------------------------------------
// MI455X (gfx1250, wave32) attention block.
// Precision strategy: bf16 WMMA (v_wmma_f32_16x16x32_bf16) with f32 accum.
// GEMMs: 128x128 block tile, K-tile 64, double-buffered LDS, async-to-LDS
// staging via global_load_async_to_lds_b128 (ASYNCcnt) when available.
// ---------------------------------------------------------------------------

#define DIMN 2048
#define NHEADS 16
#define HEADD 128
#define BB 2
#define SS 2048
#define MROWS (BB * SS)  // 4096

#define LDSK 64          // K elements staged per pipeline stage
#define LDST 72          // padded LDS row stride (elements; 144B, 16B-aligned)

typedef __attribute__((ext_vector_type(16))) __bf16 v16bf;
typedef __attribute__((ext_vector_type(8)))  float  v8f;
typedef __attribute__((ext_vector_type(4)))  unsigned int u32x4;
typedef int v4i32 __attribute__((vector_size(16)));   // builtin payload type

#if defined(__gfx1250__) && \
    __has_builtin(__builtin_amdgcn_global_load_async_to_lds_b128) && \
    __has_builtin(__builtin_amdgcn_s_wait_asynccnt)
#define HAVE_ASYNC_LDS 1
#else
#define HAVE_ASYNC_LDS 0
#endif

union Frag16 {            // 16 bf16 per lane = one WMMA A/B operand
    u32x4 u[2];
    v16bf v;
};

__device__ __forceinline__ unsigned short f2bf(float f) {
    union { float f; unsigned int u; } v; v.f = f;
    unsigned int u = v.u;
    unsigned int r = u + 0x7FFFu + ((u >> 16) & 1u);      // round-nearest-even
    if ((u & 0x7FFFFFFFu) > 0x7F800000u) r = u | 0x00400000u; // quiet NaN
    return (unsigned short)(r >> 16);
}
__device__ __forceinline__ float bf2f(unsigned short h) {
    union { unsigned int u; float f; } v; v.u = ((unsigned int)h) << 16; return v.f;
}

// Stage 32 bf16 (64B) from global into LDS: async DMA if available,
// otherwise load-to-VGPR + ds_store.
__device__ __forceinline__ void stage32(const unsigned short* __restrict__ g,
                                        unsigned short* l) {
#if HAVE_ASYNC_LDS
#pragma unroll
    for (int q = 0; q < 4; ++q)
        __builtin_amdgcn_global_load_async_to_lds_b128(
            (v4i32*)(g + q * 8), (v4i32*)(l + q * 8), 0, 0);
#else
#pragma unroll
    for (int q = 0; q < 4; ++q)
        *(u32x4*)(l + q * 8) = *(const u32x4*)(g + q * 8);
#endif
}

__device__ __forceinline__ void wait_stage() {
#if HAVE_ASYNC_LDS
    __builtin_amdgcn_s_wait_asynccnt(0);
#endif
}

// ---------------------------------------------------------------------------
// Elementwise conversion: f32 -> bf16
// ---------------------------------------------------------------------------
__global__ void cvt_f32_bf16_kernel(const float* __restrict__ in,
                                    unsigned short* __restrict__ out, int n) {
    int i = blockIdx.x * blockDim.x + threadIdx.x;
    if (i < n) out[i] = f2bf(in[i]);
}

// ---------------------------------------------------------------------------
// Block-tiled GEMM:  C[M][N] = A[M][K](bf16) * W[N][K](bf16)^T + bias
// Block = 256 thr (8 waves) -> 128x128 tile; wave = 64x32 (4x2 WMMA tiles).
// Double-buffered LDS (2 x 128x64 per operand), 2 WMMA K-steps per stage.
// ---------------------------------------------------------------------------
__global__ __launch_bounds__(256) void gemm_bf16_kernel(
    const unsigned short* __restrict__ A,   // [M][K]
    const unsigned short* __restrict__ W,   // [N][K]
    const float* __restrict__ bias,         // [N]
    float* __restrict__ outF,               // used if !bf16out
    unsigned short* __restrict__ outB,      // used if bf16out
    int M, int N, int K, int bf16out)
{
    __shared__ unsigned short lA[2][128 * LDST];
    __shared__ unsigned short lB[2][128 * LDST];

    const int tid  = threadIdx.x;
    const int wave = tid >> 5;
    const int lane = tid & 31;
    const int half = lane >> 4;
    const int l16  = lane & 15;
    const int wm   = wave >> 2;   // 0..1
    const int wn   = wave & 3;    // 0..3
    const int blockM = blockIdx.y * 128;
    const int blockN = blockIdx.x * 128;

    // each thread stages 32 bf16 (64B) of A and of W per stage
    const int ldRow = tid >> 1;
    const int ldCol = (tid & 1) * 32;

    const unsigned short* gA = A + (size_t)(blockM + ldRow) * K + ldCol;
    const unsigned short* gW = W + (size_t)(blockN + ldRow) * K + ldCol;
    const int ldsOff = ldRow * LDST + ldCol;

    v8f acc[4][2] = {};
    const int T = K / LDSK;

    // prologue: stage tile 0 into buffer 0
    stage32(gA, &lA[0][ldsOff]);
    stage32(gW, &lB[0][ldsOff]);

    for (int t = 0; t < T; ++t) {
        const int buf = t & 1;
        wait_stage();          // this wave's async copies into buf complete
        __syncthreads();       // all waves' copies into buf visible

        if (t + 1 < T) {       // overlap: stage next tile into the other buf
            stage32(gA + (t + 1) * LDSK, &lA[buf ^ 1][ldsOff]);
            stage32(gW + (t + 1) * LDSK, &lB[buf ^ 1][ldsOff]);
        }
        if (t + 2 < T) {       // gfx1250 global_prefetch_b8 two stages ahead
            __builtin_prefetch((const void*)(gA + (t + 2) * LDSK), 0, 1);
            __builtin_prefetch((const void*)(gW + (t + 2) * LDSK), 0, 1);
        }

#pragma unroll
        for (int kk = 0; kk < 2; ++kk) {
            Frag16 af[4];
#pragma unroll
            for (int i = 0; i < 4; ++i) {
                const unsigned short* p =
                    &lA[buf][(wm * 64 + i * 16 + l16) * LDST + kk * 32];
                af[i].u[0] = *(const u32x4*)(p + half * 8);
                af[i].u[1] = *(const u32x4*)(p + 16 + half * 8);
            }
            Frag16 bfr[2];
#pragma unroll
            for (int j = 0; j < 2; ++j) {
                const unsigned short* p =
                    &lB[buf][(wn * 32 + j * 16 + l16) * LDST + kk * 32 + half * 8];
                bfr[j].u[0] = *(const u32x4*)(p);
                bfr[j].u[1] = *(const u32x4*)(p + 16);
            }
#pragma unroll
            for (int i = 0; i < 4; ++i)
#pragma unroll
                for (int j = 0; j < 2; ++j)
                    acc[i][j] = __builtin_amdgcn_wmma_f32_16x16x32_bf16(
                        false, af[i].v, false, bfr[j].v, (short)0, acc[i][j],
                        false, false);
        }
    }

    // epilogue: C layout -> lane owns column (l16), VGPR r -> row half*8+r
#pragma unroll
    for (int i = 0; i < 4; ++i) {
#pragma unroll
        for (int j = 0; j < 2; ++j) {
            int row0 = blockM + wm * 64 + i * 16 + half * 8;
            int col  = blockN + wn * 32 + j * 16 + l16;
            float bv = bias[col];
#pragma unroll
            for (int r = 0; r < 8; ++r) {
                float v = acc[i][j][r] + bv;
                size_t idx = (size_t)(row0 + r) * N + col;
                if (bf16out) outB[idx] = f2bf(v);
                else         outF[idx] = v;
            }
        }
    }
}

// ---------------------------------------------------------------------------
// RoPE + repack q,k:  [b*S][DIM] bf16 -> per-head [b][h][s][128] bf16
// ---------------------------------------------------------------------------
__global__ void rope_pack_kernel(const unsigned short* __restrict__ qin,
                                 const unsigned short* __restrict__ kin,
                                 const float* __restrict__ cosp,
                                 const float* __restrict__ sinp,
                                 unsigned short* __restrict__ qout,
                                 unsigned short* __restrict__ kout, int total)
{
    int idx = blockIdx.x * blockDim.x + threadIdx.x;
    if (idx >= total) return;                // total = B*S*(DIM/2)
    int i = idx & 1023;                      // rotary pair index
    int s = (idx >> 10) & (SS - 1);
    int b = idx >> 21;
    int row = b * SS + s;
    int e0 = 2 * i;
    float c  = cosp[s * 1024 + i];
    float sn = sinp[s * 1024 + i];

    float qe = bf2f(qin[(size_t)row * DIMN + e0]);
    float qo = bf2f(qin[(size_t)row * DIMN + e0 + 1]);
    float ke = bf2f(kin[(size_t)row * DIMN + e0]);
    float ko = bf2f(kin[(size_t)row * DIMN + e0 + 1]);

    float qoe = qe * c - qo * sn, qoo = qo * c + qe * sn;
    float koe = ke * c - ko * sn, koo = ko * c + ke * sn;

    int h = e0 >> 7, d = e0 & 127;
    size_t o = (((size_t)(b * NHEADS + h) * SS) + s) * HEADD + d;
    qout[o] = f2bf(qoe); qout[o + 1] = f2bf(qoo);
    kout[o] = f2bf(koe); kout[o + 1] = f2bf(koo);
}

// V repack + transpose: [b*S][DIM] bf16 -> [b][h][d][s] bf16 (contiguous in s)
__global__ void vpack_kernel(const unsigned short* __restrict__ vin,
                             unsigned short* __restrict__ vT, int total)
{
    int idx = blockIdx.x * blockDim.x + threadIdx.x;
    if (idx >= total) return;                // total = B*S*DIM
    int e = idx & (DIMN - 1);
    int s = (idx >> 11) & (SS - 1);
    int b = idx >> 22;
    int h = e >> 7, d = e & 127;
    vT[(((size_t)(b * NHEADS + h) * HEADD) + d) * SS + s] = vin[idx];
}

// ---------------------------------------------------------------------------
// Flash attention: block = 8 waves; each wave owns 16 q-rows of one (b,h).
// Online softmax in f32; P routed through a per-wave LDS tile to convert
// C-layout scores into an A-layout operand for the P*V WMMA.
// ---------------------------------------------------------------------------
__global__ __launch_bounds__(256) void flash_attn_kernel(
    const unsigned short* __restrict__ Q,   // [B][H][S][128]
    const unsigned short* __restrict__ Kp,  // [B][H][S][128]
    const unsigned short* __restrict__ Vt,  // [B][H][128][S]
    const float* __restrict__ mask,         // [S][S]
    unsigned short* __restrict__ Out)       // [B][S][DIM] (b,s,h*128+d)
{
    __shared__ unsigned short lP[8][16 * 32];   // per-wave P tile (16x32 bf16)

    const int tid  = threadIdx.x;
    const int wave = tid >> 5;
    const int lane = tid & 31;
    const int half = lane >> 4;
    const int l16  = lane & 15;
    const int b = blockIdx.z, h = blockIdx.y;
    const int qbase = blockIdx.x * 128 + wave * 16;
    const float scale = 0.08838834764831845f;   // 1/sqrt(128)

    const unsigned short* qrow =
        Q + (((size_t)(b * NHEADS + h) * SS) + qbase + l16) * HEADD;
    Frag16 aq[4];
#pragma unroll
    for (int kk = 0; kk < 4; ++kk) {
        aq[kk].u[0] = *(const u32x4*)(qrow + kk * 32 + half * 8);
        aq[kk].u[1] = *(const u32x4*)(qrow + kk * 32 + 16 + half * 8);
    }

    const unsigned short* kbase = Kp + ((size_t)(b * NHEADS + h) * SS) * HEADD;
    const unsigned short* vbase = Vt + ((size_t)(b * NHEADS + h) * HEADD) * SS;

    v8f accO[8] = {};
    float mi[8], li[8];
#pragma unroll
    for (int r = 0; r < 8; ++r) { mi[r] = -__builtin_inff(); li[r] = 0.f; }

    const int qrowg = qbase + half * 8;

    for (int kc = 0; kc < SS; kc += 32) {
        // ---- scores S = Q K^T : two 16-col tiles, K-dim = 128 = 4 steps ----
        v8f s0 = {}, s1 = {};
        const unsigned short* kr0 = kbase + (size_t)(kc + l16) * HEADD + half * 8;
        const unsigned short* kr1 = kbase + (size_t)(kc + 16 + l16) * HEADD + half * 8;
#pragma unroll
        for (int kk = 0; kk < 4; ++kk) {
            Frag16 b0, b1;
            b0.u[0] = *(const u32x4*)(kr0 + kk * 32);
            b0.u[1] = *(const u32x4*)(kr0 + kk * 32 + 16);
            b1.u[0] = *(const u32x4*)(kr1 + kk * 32);
            b1.u[1] = *(const u32x4*)(kr1 + kk * 32 + 16);
            s0 = __builtin_amdgcn_wmma_f32_16x16x32_bf16(
                false, aq[kk].v, false, b0.v, (short)0, s0, false, false);
            s1 = __builtin_amdgcn_wmma_f32_16x16x32_bf16(
                false, aq[kk].v, false, b1.v, (short)0, s1, false, false);
        }

        // ---- scale + mask + online softmax (rows live across 16 lanes) ----
        float cmax[8];
#pragma unroll
        for (int r = 0; r < 8; ++r) {
            float m0 = mask[(size_t)(qrowg + r) * SS + kc + l16];
            float m1 = mask[(size_t)(qrowg + r) * SS + kc + 16 + l16];
            s0[r] = s0[r] * scale + m0;
            s1[r] = s1[r] * scale + m1;
            cmax[r] = fmaxf(s0[r], s1[r]);
        }
#pragma unroll
        for (int off = 8; off >= 1; off >>= 1)
#pragma unroll
            for (int r = 0; r < 8; ++r)
                cmax[r] = fmaxf(cmax[r], __shfl_xor(cmax[r], off, 32));

        float alpha[8], rsum[8];
#pragma unroll
        for (int r = 0; r < 8; ++r) {
            float mn = fmaxf(mi[r], cmax[r]);
            alpha[r] = __expf(mi[r] - mn);
            mi[r] = mn;
            s0[r] = __expf(s0[r] - mn);
            s1[r] = __expf(s1[r] - mn);
            rsum[r] = s0[r] + s1[r];
        }
#pragma unroll
        for (int off = 8; off >= 1; off >>= 1)
#pragma unroll
            for (int r = 0; r < 8; ++r)
                rsum[r] += __shfl_xor(rsum[r], off, 32);
#pragma unroll
        for (int r = 0; r < 8; ++r) li[r] = li[r] * alpha[r] + rsum[r];

        // rescale running O
#pragma unroll
        for (int j = 0; j < 8; ++j)
#pragma unroll
            for (int r = 0; r < 8; ++r) accO[j][r] *= alpha[r];

        // ---- P: C-layout -> LDS -> A-layout fragment ----
        unsigned short* p = lP[wave];
#pragma unroll
        for (int r = 0; r < 8; ++r) {
            int prow = half * 8 + r;
            p[prow * 32 + l16]      = f2bf(s0[r]);
            p[prow * 32 + 16 + l16] = f2bf(s1[r]);
        }
        asm volatile("" ::: "memory");   // keep LDS reads after writes
        Frag16 ap;
        {
            const unsigned short* pr = p + l16 * 32;
            ap.u[0] = *(const u32x4*)(pr + half * 8);
            ap.u[1] = *(const u32x4*)(pr + 16 + half * 8);
        }

        // ---- O += P * V  (K-dim = 32 k-rows, 8 d-tiles) ----
#pragma unroll
        for (int j = 0; j < 8; ++j) {
            const unsigned short* vr =
                vbase + (size_t)(j * 16 + l16) * SS + kc + half * 8;
            Frag16 bv;
            bv.u[0] = *(const u32x4*)(vr);
            bv.u[1] = *(const u32x4*)(vr + 16);
            accO[j] = __builtin_amdgcn_wmma_f32_16x16x32_bf16(
                false, ap.v, false, bv.v, (short)0, accO[j], false, false);
        }
    }

    // ---- normalize + store (B,S,H*128+d) as bf16 for the O-projection ----
#pragma unroll
    for (int j = 0; j < 8; ++j) {
#pragma unroll
        for (int r = 0; r < 8; ++r) {
            float o = accO[j][r] / li[r];
            int srow = qbase + half * 8 + r;
            size_t idx = ((size_t)b * SS + srow) * DIMN + h * HEADD + j * 16 + l16;
            Out[idx] = f2bf(o);
        }
    }
}

// ---------------------------------------------------------------------------
// Host-side orchestration (all on `stream`, graph-capture safe).
// Workspace budget ~144 MB (attention output aliases the raw-q buffer).
// ---------------------------------------------------------------------------
extern "C" void kernel_launch(void* const* d_in, const int* in_sizes, int n_in,
                              void* d_out, int out_size, void* d_ws, size_t ws_size,
                              hipStream_t stream) {
    const float* x    = (const float*)d_in[0];
    const float* cosp = (const float*)d_in[1];
    const float* sinp = (const float*)d_in[2];
    const float* mask = (const float*)d_in[3];
    const float* Wq   = (const float*)d_in[4];
    const float* bq   = (const float*)d_in[5];
    const float* Wk   = (const float*)d_in[6];
    const float* bk   = (const float*)d_in[7];
    const float* Wv   = (const float*)d_in[8];
    const float* bv   = (const float*)d_in[9];
    const float* Wo   = (const float*)d_in[10];
    const float* bo   = (const float*)d_in[11];
    float* out = (float*)d_out;

    char* ws = (char*)d_ws;
    size_t off = 0;
    auto alloc = [&](size_t bytes) -> void* {
        void* p = ws + off;
        off += (bytes + 255) & ~(size_t)255;
        return p;
    };
    const size_t MN = (size_t)MROWS * DIMN;       // 8,388,608
    const size_t WW = (size_t)DIMN * DIMN;        // 4,194,304

    unsigned short* xb   = (unsigned short*)alloc(MN * 2);
    unsigned short* Wqb  = (unsigned short*)alloc(WW * 2);
    unsigned short* Wkb  = (unsigned short*)alloc(WW * 2);
    unsigned short* Wvb  = (unsigned short*)alloc(WW * 2);
    unsigned short* Wob  = (unsigned short*)alloc(WW * 2);
    unsigned short* qg   = (unsigned short*)alloc(MN * 2);   // raw q (pre-RoPE)
    unsigned short* kg   = (unsigned short*)alloc(MN * 2);
    unsigned short* vg   = (unsigned short*)alloc(MN * 2);
    unsigned short* qP   = (unsigned short*)alloc(MN * 2);   // [b][h][s][d]
    unsigned short* kP   = (unsigned short*)alloc(MN * 2);
    unsigned short* vT   = (unsigned short*)alloc(MN * 2);   // [b][h][d][s]
    unsigned short* attn = qg;   // reuse: qg dead after RoPE repack

    // 1. fp32 -> bf16 conversions
    cvt_f32_bf16_kernel<<<(int)((MN + 255) / 256), 256, 0, stream>>>(x, xb, (int)MN);
    cvt_f32_bf16_kernel<<<(int)((WW + 255) / 256), 256, 0, stream>>>(Wq, Wqb, (int)WW);
    cvt_f32_bf16_kernel<<<(int)((WW + 255) / 256), 256, 0, stream>>>(Wk, Wkb, (int)WW);
    cvt_f32_bf16_kernel<<<(int)((WW + 255) / 256), 256, 0, stream>>>(Wv, Wvb, (int)WW);
    cvt_f32_bf16_kernel<<<(int)((WW + 255) / 256), 256, 0, stream>>>(Wo, Wob, (int)WW);

    // 2. Q/K/V projections (WMMA GEMM, bf16 out)
    dim3 ggrid(DIMN / 128, MROWS / 128);
    gemm_bf16_kernel<<<ggrid, 256, 0, stream>>>(xb, Wqb, bq, nullptr, qg,
                                                MROWS, DIMN, DIMN, 1);
    gemm_bf16_kernel<<<ggrid, 256, 0, stream>>>(xb, Wkb, bk, nullptr, kg,
                                                MROWS, DIMN, DIMN, 1);
    gemm_bf16_kernel<<<ggrid, 256, 0, stream>>>(xb, Wvb, bv, nullptr, vg,
                                                MROWS, DIMN, DIMN, 1);

    // 3. RoPE + per-head repack, V transpose
    int ropeN = BB * SS * (DIMN / 2);
    rope_pack_kernel<<<(ropeN + 255) / 256, 256, 0, stream>>>(
        qg, kg, cosp, sinp, qP, kP, ropeN);
    int vpN = BB * SS * DIMN;
    vpack_kernel<<<(vpN + 255) / 256, 256, 0, stream>>>(vg, vT, vpN);

    // 4. Flash attention (512 workgroups x 8 waves)
    dim3 agrid(SS / 128, NHEADS, BB);
    flash_attn_kernel<<<agrid, 256, 0, stream>>>(qP, kP, vT, mask, attn);

    // 5. Output projection (f32 out to d_out)
    gemm_bf16_kernel<<<ggrid, 256, 0, stream>>>(attn, Wob, bo, out, nullptr,
                                                MROWS, DIMN, DIMN, 0);
}
